// DeepLSTM_18537078850249
// MI455X (gfx1250) — matrix operand order repaired and checked by reading, not compiled, not run on hardware
//
#include <hip/hip_runtime.h>

typedef __attribute__((ext_vector_type(16))) _Float16 v16h;
typedef __attribute__((ext_vector_type(8)))  float    v8f;

#define T_LEN 1024
#define NT    2      // 16-row batch tiles per wave
#define XC    64     // timesteps per async x chunk
#define OUT_SZ 5

union HU { v16h h; unsigned int u[8]; };

__device__ __forceinline__ float tanh_(float x) {
#if __has_builtin(__builtin_amdgcn_tanhf)
  return __builtin_amdgcn_tanhf(x);
#else
  float xc = fminf(fmaxf(x, -15.0f), 15.0f);
  float e  = __expf(-2.0f * xc);
  return (1.0f - e) * __builtin_amdgcn_rcpf(1.0f + e);
#endif
}

// sigmoid via native tanh: 1/(1+e^-x) = 0.5*tanh(x/2) + 0.5
__device__ __forceinline__ float sigm_(float x) {
  return fmaf(0.5f, tanh_(0.5f * x), 0.5f);
}

// async DMA: 32 lanes x 16B = one 512B row (XC steps * 2 floats) straight to LDS
__device__ __forceinline__ void async_row_b128(const float* gsrc, const float* ldst) {
  asm volatile("global_load_async_to_lds_b128 %0, %1, off"
               :: "v"((unsigned)(size_t)ldst), "v"(gsrc) : "memory");
}

__device__ __forceinline__ void issue_chunk(const float* __restrict__ xg,
                                            const float* lbase,
                                            int b0, int t0, int lane) {
  #pragma unroll 1
  for (int r = 0; r < 32; ++r) {
    const float* g = xg + ((size_t)(b0 + r) * T_LEN + t0) * 2 + lane * 4;
    const float* l = lbase + r * (XC * 2) + lane * 4;
    async_row_b128(g, l);
  }
}

__global__ __launch_bounds__(32)
void deep_lstm_wmma(const float* __restrict__ xg,
                    const float* __restrict__ Wih1, const float* __restrict__ Whh1,
                    const float* __restrict__ bih1, const float* __restrict__ bhh1,
                    const float* __restrict__ Wih2, const float* __restrict__ Whh2,
                    const float* __restrict__ bih2, const float* __restrict__ bhh2,
                    const float* __restrict__ Wlin, const float* __restrict__ blin,
                    float* __restrict__ outp)
{
  __shared__ float    xstageF[2][32 * XC * 2];    // double-buffered raw f32 x, 2x16KB
  __shared__ _Float16 buf1[NT][16 * 32];          // per-row [x(2)|h1(16)|pad(14)]
  __shared__ _Float16 buf2[NT][16 * 32];          // per-row [x(2)|h2(16)|pad(14)]

  const int lane  = threadIdx.x;
  const int n     = lane & 15;          // N column (C layout) / A row (A layout)
  const int hi    = lane >> 4;          // lane half
  const int kbase = hi ? 8 : 0;         // B layout K base for this lane half
  const int b0    = blockIdx.x * (16 * NT);

  // ---------------- weight B-tiles (K=32 x N=16, f16) ----------------
  // B layout: VGPR j, k = 16*(j/4) + 8*hi + 2*(j%4) + {0,1} (N=lane)
  HU W1[4], W2a[4], W2b[4], WL;
  #pragma unroll
  for (int q = 0; q < 4; ++q) {
    const int ng = q * 16 + n;          // global gate column (i,f,g,o blocks)
    #pragma unroll
    for (int j = 0; j < 8; ++j) {
      #pragma unroll
      for (int hh = 0; hh < 2; ++hh) {
        const int k = kbase + 2 * j + hh + 8 * (j >> 2);
        const float f1  = (k < 2) ? Wih1[ng * 2 + k]
                        : ((k < 18) ? Whh1[ng * 16 + (k - 2)] : 0.0f);
        const float f2a = (k < 18) ? Wih2[ng * 18 + k] : 0.0f;                 // [x|h1n]
        const float f2b = (k >= 2 && k < 18) ? Whh2[ng * 16 + (k - 2)] : 0.0f; // h2 slot
        W1[q].h[2 * j + hh]  = (_Float16)f1;
        W2a[q].h[2 * j + hh] = (_Float16)f2a;
        W2b[q].h[2 * j + hh] = (_Float16)f2b;
      }
    }
  }
  #pragma unroll
  for (int j = 0; j < 8; ++j) {
    #pragma unroll
    for (int hh = 0; hh < 2; ++hh) {
      const int k = kbase + 2 * j + hh + 8 * (j >> 2);
      const float f = (n < OUT_SZ && k < 18) ? Wlin[n * 18 + k] : 0.0f;
      WL.h[2 * j + hh] = (_Float16)f;
    }
  }

  float bias1[4], bias2[4];
  #pragma unroll
  for (int q = 0; q < 4; ++q) {
    bias1[q] = bih1[q * 16 + n] + bhh1[q * 16 + n];
    bias2[q] = bih2[q * 16 + n] + bhh2[q * 16 + n];
  }
  const float blv = (n < OUT_SZ) ? blin[n] : 0.0f;

  // zero staging buffers: h regions start at 0, pad columns stay 0 forever
  for (int idx = lane; idx < NT * 16 * 16; idx += 32) {
    ((unsigned int*)buf1)[idx] = 0u;
    ((unsigned int*)buf2)[idx] = 0u;
  }
  __syncthreads();

  float c1[NT][8], c2[NT][8];
  #pragma unroll
  for (int tt = 0; tt < NT; ++tt)
    #pragma unroll
    for (int e = 0; e < 8; ++e) { c1[tt][e] = 0.0f; c2[tt][e] = 0.0f; }

  // prologue: kick off async DMA of chunk 0
  issue_chunk(xg, xstageF[0], b0, 0, lane);

  const int NCHUNK = T_LEN / XC;
  for (int ch = 0; ch < NCHUNK; ++ch) {
    const int t0 = ch * XC;

    if (ch + 1 < NCHUNK) {
      // overlap: issue next chunk's DMA, then wait for current (allow 32 in flight)
      issue_chunk(xg, xstageF[(ch + 1) & 1], b0, t0 + XC, lane);
      asm volatile("s_wait_asynccnt 0x20" ::: "memory");
    } else {
      asm volatile("s_wait_asynccnt 0x0" ::: "memory");
    }
    if (ch + 2 < NCHUNK)  // hint chunk+2 toward L2
      __builtin_prefetch(xg + ((size_t)(b0 + lane) * T_LEN + (t0 + 2 * XC)) * 2, 0, 1);
    __syncthreads();

    const float* xs = xstageF[ch & 1];

    #pragma unroll 1
    for (int jj = 0; jj < XC; ++jj) {
      const int t = t0 + jj;

      { // place x_t into cols 0-1 of both buffers (one row per lane)
        const int tt = lane >> 4, m = lane & 15;
        const float2 xv = *(const float2*)(xs + (lane * XC + jj) * 2);
        const auto pk = __builtin_amdgcn_cvt_pkrtz(xv.x, xv.y);  // __fp16 x2
        const unsigned int xu = __builtin_bit_cast(unsigned int, pk);
        ((unsigned int*)buf1[tt])[m * 16] = xu;
        ((unsigned int*)buf2[tt])[m * 16] = xu;
      }
      __syncthreads();

      // ------------------------------ LSTM cell 1 ------------------------------
      #pragma unroll
      for (int tt = 0; tt < NT; ++tt) {
        HU a;   // A layout: row=n, k = 16*(j/4) + 8*hi + 2*(j%4)
        #pragma unroll
        for (int j = 0; j < 8; ++j) {
          const int ku = 8 * (j >> 2) + 4 * hi + (j & 3);
          a.u[j] = ((const unsigned int*)buf1[tt])[n * 16 + ku];
        }
        v8f g[4];
        #pragma unroll
        for (int q = 0; q < 4; ++q) {
          v8f c;
          #pragma unroll
          for (int e = 0; e < 8; ++e) c[e] = bias1[q];
          g[q] = __builtin_amdgcn_wmma_f32_16x16x32_f16(false, a.h, false, W1[q].h,
                                                        (short)0, c, false, false);
        }
        #pragma unroll
        for (int e = 0; e < 8; ++e) {
          const float ig = sigm_(g[0][e]);
          const float fg = sigm_(g[1][e]);
          const float gg = tanh_(g[2][e]);
          const float og = sigm_(g[3][e]);
          const float cn = fmaf(fg, c1[tt][e], ig * gg);
          c1[tt][e] = cn;
          const float hn = og * tanh_(cn);
          buf1[tt][(e + 8 * hi) * 32 + 2 + n] = (_Float16)hn;   // h1n into buf1
        }
      }
      __syncthreads();

      // ------------------------------ LSTM cell 2 ------------------------------
      #pragma unroll
      for (int tt = 0; tt < NT; ++tt) {
        HU a, b2;
        #pragma unroll
        for (int j = 0; j < 8; ++j) {
          const int ku = 8 * (j >> 2) + 4 * hi + (j & 3);
          a.u[j]  = ((const unsigned int*)buf1[tt])[n * 16 + ku];  // [x|h1n]
          b2.u[j] = ((const unsigned int*)buf2[tt])[n * 16 + ku];  // [x|h2] (x rows 0 in W2b)
        }
        v8f g[4];
        #pragma unroll
        for (int q = 0; q < 4; ++q) {
          v8f c;
          #pragma unroll
          for (int e = 0; e < 8; ++e) c[e] = bias2[q];
          c    = __builtin_amdgcn_wmma_f32_16x16x32_f16(false, a.h,  false, W2a[q].h,
                                                        (short)0, c, false, false);
          g[q] = __builtin_amdgcn_wmma_f32_16x16x32_f16(false, b2.h, false, W2b[q].h,
                                                        (short)0, c, false, false);
        }
        #pragma unroll
        for (int e = 0; e < 8; ++e) {
          const float ig = sigm_(g[0][e]);
          const float fg = sigm_(g[1][e]);
          const float gg = tanh_(g[2][e]);
          const float og = sigm_(g[3][e]);
          const float cn = fmaf(fg, c2[tt][e], ig * gg);
          c2[tt][e] = cn;
          const float hn = og * tanh_(cn);
          buf2[tt][(e + 8 * hi) * 32 + 2 + n] = (_Float16)hn;   // h2n into buf2
        }
      }
      __syncthreads();

      // ------------------------------ output linear ----------------------------
      #pragma unroll
      for (int tt = 0; tt < NT; ++tt) {
        HU a;
        #pragma unroll
        for (int j = 0; j < 8; ++j) {
          const int ku = 8 * (j >> 2) + 4 * hi + (j & 3);
          a.u[j] = ((const unsigned int*)buf2[tt])[n * 16 + ku];  // [x|h2n]
        }
        v8f y;
        #pragma unroll
        for (int e = 0; e < 8; ++e) y[e] = blv;
        y = __builtin_amdgcn_wmma_f32_16x16x32_f16(false, a.h, false, WL.h,
                                                   (short)0, y, false, false);
        if (n < OUT_SZ) {
          #pragma unroll
          for (int e = 0; e < 8; ++e) {
            const int m = e + 8 * hi;
            outp[((size_t)(b0 + tt * 16 + m) * T_LEN + t) * OUT_SZ + n] = y[e];
          }
        }
      }
      __syncthreads();
    }
  }
}

extern "C" void kernel_launch(void* const* d_in, const int* in_sizes, int n_in,
                              void* d_out, int out_size, void* d_ws, size_t ws_size,
                              hipStream_t stream) {
  const float* x    = (const float*)d_in[0];
  const float* Wih1 = (const float*)d_in[1];
  const float* Whh1 = (const float*)d_in[2];
  const float* bih1 = (const float*)d_in[3];
  const float* bhh1 = (const float*)d_in[4];
  const float* Wih2 = (const float*)d_in[5];
  const float* Whh2 = (const float*)d_in[6];
  const float* bih2 = (const float*)d_in[7];
  const float* bhh2 = (const float*)d_in[8];
  const float* Wl   = (const float*)d_in[9];
  const float* bl   = (const float*)d_in[10];
  float* out = (float*)d_out;

  const int grid = 4096 / (16 * NT);   // 128 single-wave workgroups
  deep_lstm_wmma<<<grid, 32, 0, stream>>>(x, Wih1, Whh1, bih1, bhh1,
                                          Wih2, Whh2, bih2, bhh2, Wl, bl, out);
}
